// LSTM_2302102471468
// MI455X (gfx1250) — matrix operand order repaired and checked by reading, not compiled, not run on hardware
//
#include <hip/hip_runtime.h>

// Problem sizes (match reference)
static constexpr int Bsz = 64;
static constexpr int Ssz = 512;
static constexpr int Dsz = 256;
static constexpr int Hsz = 512;
static constexpr int Gsz = 4 * Hsz;   // 2048
static constexpr int NBLK = 32;       // persistent grid: 32 blocks x 4 waves = 128 wave-tasks
static constexpr int KXS = Dsz / 32;  // 8 k-steps for input projection
static constexpr int KHS = Hsz / 32;  // 16 k-steps for recurrence

typedef __attribute__((ext_vector_type(16))) __bf16 v16bf;
typedef __attribute__((ext_vector_type(8)))  float  v8f;

union Frag32 { uint4 u[2]; v16bf v; };

__device__ __forceinline__ unsigned short f2bf(float f) {
  unsigned u = __float_as_uint(f);
  unsigned r = (u + 0x7FFFu + ((u >> 16) & 1u)) >> 16;  // round-to-nearest-even
  return (unsigned short)r;
}

// ---------------------------------------------------------------------------
// Prep: convert inp / W_ih / W_hh to bf16, bias = b_ih + b_hh, zero h/barrier
// ---------------------------------------------------------------------------
__global__ void lstm_prep(const float* __restrict__ inp,
                          const float* __restrict__ W_ih,
                          const float* __restrict__ W_hh,
                          const float* __restrict__ b_ih,
                          const float* __restrict__ b_hh,
                          unsigned short* __restrict__ inp_b,
                          unsigned short* __restrict__ Wih_b,
                          unsigned short* __restrict__ Whh_b,
                          float* __restrict__ bias,
                          unsigned short* __restrict__ hbuf,
                          unsigned* __restrict__ bar) {
  int idx = blockIdx.x * blockDim.x + threadIdx.x;
  if (idx < Bsz * Ssz * Dsz) inp_b[idx] = f2bf(inp[idx]);
  if (idx < Gsz * Dsz)       Wih_b[idx] = f2bf(W_ih[idx]);
  if (idx < Gsz * Hsz)       Whh_b[idx] = f2bf(W_hh[idx]);
  if (idx < Gsz)             bias[idx]  = b_ih[idx] + b_hh[idx];
  if (idx < 2 * Bsz * Hsz)   hbuf[idx]  = 0;
  if (idx == 0)              *bar       = 0u;
}

// ---------------------------------------------------------------------------
// Persistent register-resident LSTM: 32 blocks x 4 waves, 1 wave/SIMD.
// Weights (768 VGPRs) + c-state live in registers for all 512 steps.
// h(t-1) is staged block-wide into 64KB of XOR-swizzled LDS once per step;
// x(t+1) is prefetched into the WGP cache one step ahead.
// ---------------------------------------------------------------------------
__global__ void __launch_bounds__(128, 1)
__attribute__((amdgpu_waves_per_eu(1, 1)))
lstm_persistent(const unsigned short* __restrict__ inp_b,
                const unsigned short* __restrict__ Wih_b,
                const unsigned short* __restrict__ Whh_b,
                const float* __restrict__ bias,
                const int* __restrict__ length,
                unsigned short* __restrict__ hbuf,
                float* __restrict__ out,
                unsigned* __restrict__ bar) {
  // 64 rows x 64 chunks of 16B = exactly 64 KB (h matrix, swizzled)
  __shared__ uint4 sH4[Bsz * (Hsz / 8)];

  const int lane  = threadIdx.x & 31;
  const int wave  = threadIdx.x >> 5;     // bTile 0..3
  const int bBase = wave * 16;
  const int hBase = blockIdx.x * 16;      // hTile 0..31
  const int col   = lane & 15;
  const int hi    = lane >> 4;            // half-wave select
  const int aLo   = hi * 8;               // unused; A frags also use K offset hi*16 (elems)
  const int bLo   = hi * 16;              // B-frag K chunk offset (elems)

  // Per-column bias per gate
  float bv[4];
#pragma unroll
  for (int g = 0; g < 4; ++g) bv[g] = bias[g * Hsz + hBase + col];

  // ---- preload ALL weight fragments into registers (persistent-RNN) ----
  Frag32 wih[4][KXS];   // 4*8*32B  = 256 VGPRs
  Frag32 whh[4][KHS];   // 4*16*32B = 512 VGPRs
#pragma unroll
  for (int g = 0; g < 4; ++g) {
    const unsigned short* r = Wih_b + (size_t)(g * Hsz + hBase + col) * Dsz;
#pragma unroll
    for (int ks = 0; ks < KXS; ++ks) {
      wih[g][ks].u[0] = *(const uint4*)(r + ks * 32 + bLo);
      wih[g][ks].u[1] = *(const uint4*)(r + ks * 32 + bLo + 8);
    }
  }
#pragma unroll
  for (int g = 0; g < 4; ++g) {
    const unsigned short* r = Whh_b + (size_t)(g * Hsz + hBase + col) * Hsz;
#pragma unroll
    for (int ks = 0; ks < KHS; ++ks) {
      whh[g][ks].u[0] = *(const uint4*)(r + ks * 32 + bLo);
      whh[g][ks].u[1] = *(const uint4*)(r + ks * 32 + bLo + 8);
    }
  }

  // Final-timestep index per owned batch row
  int lenm1[8];
#pragma unroll
  for (int j = 0; j < 8; ++j) lenm1[j] = length[bBase + j + hi * 8] - 1;

  const unsigned short* aX = inp_b + (size_t)(bBase + col) * (Ssz * Dsz);
  const int hcol = hBase + col;
  const int hrow = bBase + col;           // A-matrix row this lane owns
  const int swz  = (hrow & 7) << 1;       // LDS chunk swizzle
  const int rb   = hrow << 6;             // chunk base of this lane's h row

  // c-state is private to this wave: registers for all 512 steps
  float creg[8];
#pragma unroll
  for (int j = 0; j < 8; ++j) creg[j] = 0.0f;

#pragma unroll 1
  for (int t = 0; t < Ssz; ++t) {
    const unsigned short* hin  = hbuf + (size_t)(t & 1) * (Bsz * Hsz);
    unsigned short*       hout = hbuf + (size_t)((t + 1) & 1) * (Bsz * Hsz);

    // ---- cooperative stage of h(t-1): global -> swizzled LDS (64KB) ----
    {
      const uint4* hg4 = (const uint4*)hin;   // 4096 chunks
#pragma unroll 1
      for (int blk = 0; blk < 4; ++blk) {
        uint4 tmp[8];
#pragma unroll
        for (int i = 0; i < 8; ++i)
          tmp[i] = hg4[(blk * 8 + i) * 128 + threadIdx.x];
#pragma unroll
        for (int i = 0; i < 8; ++i) {
          int q = (blk * 8 + i) * 128 + threadIdx.x;
          int row = q >> 6, c8 = q & 63;
          sH4[(row << 6) | (c8 ^ ((row & 7) << 1))] = tmp[i];
        }
      }
    }

    // ---- prefetch next timestep's x row into WGP cache ----
    if (t + 1 < Ssz) {
      const char* p = (const char*)(aX + (size_t)(t + 1) * Dsz);
      __builtin_prefetch(p, 0, 3);
      __builtin_prefetch(p + 128, 0, 3);
      __builtin_prefetch(p + 256, 0, 3);
      __builtin_prefetch(p + 384, 0, 3);
    }

    __syncthreads();   // sH4 ready

    v8f acc[4];
#pragma unroll
    for (int g = 0; g < 4; ++g)
      acc[g] = (v8f){bv[g], bv[g], bv[g], bv[g], bv[g], bv[g], bv[g], bv[g]};

    // ---- input projection: K = 256, A from global (prefetched) ----
    const unsigned short* aRow = aX + (size_t)t * Dsz;
#pragma unroll
    for (int ks = 0; ks < KXS; ++ks) {
      Frag32 a;
      a.u[0] = *(const uint4*)(aRow + ks * 32 + bLo);
      a.u[1] = *(const uint4*)(aRow + ks * 32 + bLo + 8);
#pragma unroll
      for (int g = 0; g < 4; ++g)
        acc[g] = __builtin_amdgcn_wmma_f32_16x16x32_bf16(
            false, a.v, false, wih[g][ks].v, (short)0, acc[g], false, false);
    }

    // ---- recurrence: K = 512, A fragments from swizzled LDS ----
#pragma unroll
    for (int ks = 0; ks < KHS; ++ks) {
      Frag32 a;
      a.u[0] = sH4[rb | ((ks * 4 + 2 * hi) ^ swz)];
      a.u[1] = sH4[rb | ((ks * 4 + 2 * hi + 1) ^ swz)];
#pragma unroll
      for (int g = 0; g < 4; ++g)
        acc[g] = __builtin_amdgcn_wmma_f32_16x16x32_bf16(
            false, a.v, false, whh[g][ks].v, (short)0, acc[g], false, false);
    }

    // ---- LSTM cell update (all in registers) ----
#pragma unroll
    for (int j = 0; j < 8; ++j) {
      int b = bBase + j + hi * 8;             // C/D layout: VGPR j, half-wave hi
      float gi = acc[0][j], gf = acc[1][j], gg = acc[2][j], go = acc[3][j];
      float si = 1.0f / (1.0f + __expf(-gi));
      float sf = 1.0f / (1.0f + __expf(-gf));
      float so = 1.0f / (1.0f + __expf(-go));
      float cN = sf * creg[j] + si * tanhf(gg);
      float hN = so * tanhf(cN);
      creg[j] = cN;
      size_t idx = (size_t)b * Hsz + hcol;
      hout[idx] = f2bf(hN);                   // read by all blocks next step
      if (t == lenm1[j]) out[idx] = hN;       // pooling == 'final'
    }

    // ---- device-wide step barrier (32 co-resident blocks) ----
    __threadfence();
    __syncthreads();
    if (threadIdx.x == 0) {
      __hip_atomic_fetch_add(bar, 1u, __ATOMIC_RELEASE, __HIP_MEMORY_SCOPE_AGENT);
      unsigned target = (unsigned)(t + 1) * (unsigned)NBLK;
      while (__hip_atomic_load(bar, __ATOMIC_ACQUIRE, __HIP_MEMORY_SCOPE_AGENT) < target) {
        __builtin_amdgcn_s_sleep(2);
      }
    }
    __syncthreads();
  }
}

// ---------------------------------------------------------------------------
// Host launcher
// ---------------------------------------------------------------------------
extern "C" void kernel_launch(void* const* d_in, const int* in_sizes, int n_in,
                              void* d_out, int out_size, void* d_ws, size_t ws_size,
                              hipStream_t stream) {
  (void)in_sizes; (void)n_in; (void)out_size; (void)ws_size;
  const float* inp  = (const float*)d_in[0];
  const int*   len  = (const int*)  d_in[1];
  const float* W_ih = (const float*)d_in[2];
  const float* W_hh = (const float*)d_in[3];
  const float* b_ih = (const float*)d_in[4];
  const float* b_hh = (const float*)d_in[5];
  float* out = (float*)d_out;

  // Workspace layout (~20 MB, all 16B-aligned)
  char* ws = (char*)d_ws;
  size_t o = 0;
  unsigned short* inp_b = (unsigned short*)(ws + o); o += (size_t)Bsz * Ssz * Dsz * 2;
  unsigned short* Wih_b = (unsigned short*)(ws + o); o += (size_t)Gsz * Dsz * 2;
  unsigned short* Whh_b = (unsigned short*)(ws + o); o += (size_t)Gsz * Hsz * 2;
  float*          bias  = (float*)        (ws + o); o += (size_t)Gsz * 4;
  unsigned short* hbuf  = (unsigned short*)(ws + o); o += (size_t)2 * Bsz * Hsz * 2;
  unsigned*       bar   = (unsigned*)     (ws + o);

  int total = Bsz * Ssz * Dsz;  // largest conversion range
  lstm_prep<<<(total + 255) / 256, 256, 0, stream>>>(
      inp, W_ih, W_hh, b_ih, b_hh, inp_b, Wih_b, Whh_b, bias, hbuf, bar);

  lstm_persistent<<<NBLK, 128, 0, stream>>>(
      inp_b, Wih_b, Whh_b, bias, len, hbuf, out, bar);
}
